// SpatialAttention3D_71442486001740
// MI455X (gfx1250) — matrix operand order repaired and checked by reading, not compiled, not run on hardware
//
#include <hip/hip_runtime.h>
#include <hip/hip_bf16.h>
#include <stdint.h>

// ---------------------------------------------------------------------------
// SpatialAttention3D for MI455X (gfx1250, wave32, WMMA)
//   B=2, C=384, D=H=W=32, HEADS=12, HD=32, WS=5
// Pipeline:
//   prep:  w_qkv/w_proj/pos_embed fp32 -> bf16 (pe transposed to [off][hd],
//          padded to 128 offsets with zeros)
//   gemm1: QKV projection via v_wmma_f32_16x16x32_bf16 -> q,k,v bf16 [b][h][s][hd]
//   attn:  qpe = Q·PE via WMMA (per-wave, zero-waste), then streaming-softmax
//          windowed attention; OOB offsets use qpe only (k zero-padded)
//   gemm2: output projection via WMMA -> fp32 d_out [b][c][s]
// ---------------------------------------------------------------------------

typedef __attribute__((ext_vector_type(16))) __bf16          v16bf;
typedef __attribute__((ext_vector_type(8)))  float           v8f;
typedef __attribute__((ext_vector_type(8)))  unsigned short  u16x8;

#define B_      2
#define C_      384
#define S_      32768          // 32*32*32
#define HEADS_  12
#define HD_     32
#define SCALE_  0.17677669529663687f   // 32^-0.5
#define KP_     392            // padded LDS stride (bf16 elems), mult of 8, bank-friendly
#define QPEP_   130            // qpe LDS row stride: 65 dwords -> lane==bank on reads

__device__ __forceinline__ unsigned short f2bf(float f) {
  unsigned u = __float_as_uint(f);
  u += 0x7FFFu + ((u >> 16) & 1u);            // round-to-nearest-even
  return (unsigned short)(u >> 16);
}
__device__ __forceinline__ float bf2f(unsigned short h) {
  return __uint_as_float(((unsigned)h) << 16);
}
__device__ __forceinline__ v16bf mk16(u16x8 lo, u16x8 hi) {
  return __builtin_bit_cast(
      v16bf, __builtin_shufflevector(lo, hi, 0, 1, 2, 3, 4, 5, 6, 7, 8, 9, 10,
                                     11, 12, 13, 14, 15));
}

// ---------------------------------------------------------------------------
// prep: weights + pos_embed -> bf16; pe transposed to [off][hd], offs 125..127
// zero-filled so the padded WMMA B-tiles are clean.
// ---------------------------------------------------------------------------
__global__ __launch_bounds__(256) void prep_kernel(
    const float* __restrict__ wqkv, const float* __restrict__ wproj,
    const float* __restrict__ pe,
    unsigned short* __restrict__ wqkv_bf, unsigned short* __restrict__ wproj_bf,
    unsigned short* __restrict__ pe_bf)
{
  int i = blockIdx.x * 256 + threadIdx.x;
  if (i < 3 * C_ * C_) wqkv_bf[i] = f2bf(wqkv[i]);
  if (i < C_ * C_)     wproj_bf[i] = f2bf(wproj[i]);
  if (i < 128 * HD_) {                 // transpose (hd,125) -> (128,hd), pad 0
    int hd = i & 31, off = i >> 5;
    pe_bf[i] = (off < 125) ? f2bf(pe[hd * 125 + off]) : (unsigned short)0;
  }
}

// ---------------------------------------------------------------------------
// GEMM: out[o, pos] = sum_c W[o,c] * in[c,pos] + bias[o]
//   A = weight tile 16x32 (direct global b128 loads, bf16 row-major)
//   B = activation tile 32x16 staged transposed in LDS (b128 ds loads)
//   one block = 16 positions, 8 waves x NTILES o-tiles = all N output channels
// ---------------------------------------------------------------------------
template <int NTILES, bool IN_F32, bool OUT_QKV>
__global__ __launch_bounds__(256) void gemm_kernel(
    const void* __restrict__ inp, const unsigned short* __restrict__ wbf,
    const float* __restrict__ bias,
    unsigned short* __restrict__ qb, unsigned short* __restrict__ kb,
    unsigned short* __restrict__ vb, float* __restrict__ out)
{
  __shared__ unsigned short lds[16 * KP_];     // 16 pos x 384 c (padded), bf16

  const int tid = threadIdx.x;
  const int tile = blockIdx.x;                 // 0..4095 (16-position tiles)
  const int b  = tile >> 11;                   // 2048 tiles per batch
  const int s0 = (tile & 2047) << 4;

  // ---- stage activation tile [pos][c] into LDS (coalesced along pos) ----
#pragma unroll
  for (int j = 0; j < 24; ++j) {               // 16*384 = 6144 = 24*256
    int idx = tid + j * 256;
    int c = idx >> 4, n = idx & 15;
    unsigned short hv;
    if constexpr (IN_F32) {
      const float* xp = (const float*)inp;
      hv = f2bf(xp[((size_t)(b * C_ + c)) * S_ + s0 + n]);
    } else {
      const unsigned short* xp = (const unsigned short*)inp;
      hv = xp[((size_t)(b * C_ + c)) * S_ + s0 + n];
    }
    lds[n * KP_ + c] = hv;
  }
  __syncthreads();

  const int lane  = tid & 31;
  const int wv    = tid >> 5;                  // wave id 0..7
  const int nrow  = lane & 15;                 // N (position) / A-row (o)
  const int khalf = lane >> 4;

  v8f acc[NTILES];
#pragma unroll
  for (int t = 0; t < NTILES; ++t) acc[t] = (v8f){};

  for (int ks = 0; ks < 12; ++ks) {            // K = 384 = 12 * 32
    // B tile from LDS: lane needs K = khalf*16 .. +15 at row nrow
    const u16x8* bp =
        (const u16x8*)&lds[nrow * KP_ + ks * 32 + khalf * 16];
    v16bf bmat = mk16(bp[0], bp[1]);
#pragma unroll
    for (int t = 0; t < NTILES; ++t) {
      int orow = (t * 8 + wv) * 16 + nrow;     // A row (output channel)
      const unsigned short* ab = &wbf[(size_t)orow * C_ + ks * 32 + khalf * 16];
      v16bf amat = mk16(*(const u16x8*)ab, *(const u16x8*)(ab + 8));
      acc[t] = __builtin_amdgcn_wmma_f32_16x16x32_bf16(
          false, amat, false, bmat, (short)0, acc[t], false, false);
    }
  }

  // ---- epilogue: D layout lane=N(pos), VGPR r -> M = r + 8*khalf ----
#pragma unroll
  for (int t = 0; t < NTILES; ++t) {
    int otile = (t * 8 + wv) * 16;
    if constexpr (OUT_QKV) {
      int sel = otile / C_;                    // 0=q 1=k 2=v (constant per tile)
      unsigned short* dst = (sel == 0) ? qb : ((sel == 1) ? kb : vb);
#pragma unroll
      for (int r = 0; r < 8; r += 2) {
        int o0 = otile + r + 8 * khalf;        // even -> hd even -> 4B aligned
        float v0 = acc[t][r]     + bias[o0];
        float v1 = acc[t][r + 1] + bias[o0 + 1];
        int within = o0 % C_;
        int head = within >> 5, hd = within & 31;
        size_t a =
            (((size_t)(b * HEADS_ + head)) * S_ + (s0 + nrow)) * HD_ + hd;
        unsigned pk = (unsigned)f2bf(v0) | ((unsigned)f2bf(v1) << 16);
        *(unsigned*)&dst[a] = pk;
      }
    } else {
#pragma unroll
      for (int r = 0; r < 8; ++r) {
        int o = otile + r + 8 * khalf;
        out[((size_t)(b * C_ + o)) * S_ + s0 + nrow] = acc[t][r] + bias[o];
      }
    }
  }
}

// ---------------------------------------------------------------------------
// Windowed attention. Block = 128 threads = 4 waves = 128 consecutive
// positions of one (b,head). Phase 1: each wave computes qpe[pos][off] =
// SCALE * Q·PE with 16 zero-waste WMMAs into LDS. Phase 2: per-thread
// streaming softmax; OOB offsets use qpe only (k is zero-padded in the ref).
// ---------------------------------------------------------------------------
__global__ __launch_bounds__(128) void attn_kernel(
    const unsigned short* __restrict__ qb, const unsigned short* __restrict__ kb,
    const unsigned short* __restrict__ vb, const unsigned short* __restrict__ peb,
    unsigned short* __restrict__ aout)
{
  __shared__ unsigned short qpe[128 * QPEP_];  // 33280 B, bank-conflict-free

  const int tid  = threadIdx.x;
  const int gid  = blockIdx.x * 128 + tid;     // 0 .. B*HEADS*S-1
  const int s    = gid & (S_ - 1);
  const int bh   = gid >> 15;                  // b*HEADS + head (const per block)
  const int lane = tid & 31, wv = tid >> 5;
  const int nrow = lane & 15, khalf = lane >> 4;
  const int sblock = (blockIdx.x * 128) & (S_ - 1);

  // ---- phase 1: qpe = SCALE * (Q 16x32) x (PE 32x128) per wave ----
#pragma unroll
  for (int half = 0; half < 2; ++half) {
    int prow = wv * 32 + half * 16 + nrow;     // A-row local position
    const unsigned short* qrow = &qb[((size_t)bh * S_ + sblock + prow) * HD_];
    v16bf amat = mk16(*(const u16x8*)(qrow + khalf * 16),
                      *(const u16x8*)(qrow + khalf * 16 + 8));
#pragma unroll
    for (int it = 0; it < 8; ++it) {           // 128 offsets = 8 N-tiles
      const unsigned short* pb =
          &peb[(size_t)(it * 16 + nrow) * HD_ + khalf * 16];
      v16bf bmat = mk16(*(const u16x8*)pb, *(const u16x8*)(pb + 8));
      v8f d = (v8f){};
      d = __builtin_amdgcn_wmma_f32_16x16x32_bf16(
          false, amat, false, bmat, (short)0, d, false, false);
#pragma unroll
      for (int r = 0; r < 8; ++r) {
        int pl = wv * 32 + half * 16 + r + 8 * khalf;
        qpe[pl * QPEP_ + it * 16 + nrow] = f2bf(d[r] * SCALE_);
      }
    }
  }
  __syncthreads();

  // ---- phase 2: streaming softmax over 125 offsets ----
  int wq = s & 31, hq = (s >> 5) & 31, dq = s >> 10;

  float q[32];
  {
    const u16x8* qp = (const u16x8*)&qb[((size_t)bh * S_ + s) * HD_];
#pragma unroll
    for (int j = 0; j < 4; ++j) {
      u16x8 t = qp[j];
#pragma unroll
      for (int e = 0; e < 8; ++e) q[j * 8 + e] = bf2f(t[e]);
    }
  }

  float m = -1e30f, l = 0.f;
  float acc[32];
#pragma unroll
  for (int j = 0; j < 32; ++j) acc[j] = 0.f;

  int i = 0;
  for (int dz = 0; dz < 5; ++dz)
    for (int dy = 0; dy < 5; ++dy)
      for (int dx = 0; dx < 5; ++dx, ++i) {
        int zz = dq + dz - 2, yy = hq + dy - 2, xx = wq + dx - 2;
        bool inb = ((unsigned)zz < 32u) & ((unsigned)yy < 32u) &
                   ((unsigned)xx < 32u);
        float qpe_s = bf2f(qpe[tid * QPEP_ + i]);  // SCALE * q·pe_i

        float sd = qpe_s;                      // OOB: k=0 -> score = q·pe only
        size_t kbase = 0;
        if (inb) {
          int sn = (zz << 10) + (yy << 5) + xx;
          kbase = ((size_t)bh * S_ + sn) * HD_;
          const u16x8* kp = (const u16x8*)&kb[kbase];
          float dot = 0.f;
#pragma unroll
          for (int j = 0; j < 4; ++j) {
            u16x8 kv = kp[j];
#pragma unroll
            for (int e = 0; e < 8; ++e) dot += q[j * 8 + e] * bf2f(kv[e]);
          }
          sd = fmaf(dot, SCALE_, qpe_s);
        }

        float mn    = fmaxf(m, sd);
        float alpha = __expf(m - mn);
        float wgt   = __expf(sd - mn);
        m = mn;
        l = l * alpha + wgt;

        if (inb) {
          const u16x8* vp = (const u16x8*)&vb[kbase];
#pragma unroll
          for (int j = 0; j < 4; ++j) {
            u16x8 vv = vp[j];
#pragma unroll
            for (int e = 0; e < 8; ++e)
              acc[j * 8 + e] = acc[j * 8 + e] * alpha + wgt * bf2f(vv[e]);
          }
        } else {
#pragma unroll
          for (int j = 0; j < 32; ++j) acc[j] *= alpha;
        }
      }

  float inv = 1.f / l;
  int b = bh / HEADS_, head = bh % HEADS_;
  size_t ob = ((size_t)(b * C_ + head * HD_)) * S_ + s;
#pragma unroll
  for (int j = 0; j < 32; ++j) aout[ob + (size_t)j * S_] = f2bf(acc[j] * inv);
}

// ---------------------------------------------------------------------------
extern "C" void kernel_launch(void* const* d_in, const int* in_sizes, int n_in,
                              void* d_out, int out_size, void* d_ws,
                              size_t ws_size, hipStream_t stream) {
  (void)in_sizes; (void)n_in; (void)out_size; (void)ws_size;
  const float* x     = (const float*)d_in[0];
  const float* wqkv  = (const float*)d_in[1];
  const float* bqkv  = (const float*)d_in[2];
  const float* wproj = (const float*)d_in[3];
  const float* bproj = (const float*)d_in[4];
  const float* pe    = (const float*)d_in[5];

  char* ws = (char*)d_ws;
  size_t o = 0;
  unsigned short* WQ = (unsigned short*)(ws + o); o += (size_t)3 * C_ * C_ * 2;   // 884736
  unsigned short* WP = (unsigned short*)(ws + o); o += (size_t)C_ * C_ * 2;       // 294912
  unsigned short* PE = (unsigned short*)(ws + o); o += 8192;                      // 128*32*2
  const size_t QKV_BYTES = (size_t)B_ * HEADS_ * S_ * HD_ * 2;                    // 50331648
  unsigned short* QB = (unsigned short*)(ws + o); o += QKV_BYTES;
  unsigned short* KB = (unsigned short*)(ws + o); o += QKV_BYTES;
  unsigned short* VB = (unsigned short*)(ws + o); o += QKV_BYTES;
  unsigned short* AO = (unsigned short*)(ws + o); o += QKV_BYTES;

  // 1) weight / pos-embed conversion
  prep_kernel<<<(3 * C_ * C_ + 255) / 256, 256, 0, stream>>>(wqkv, wproj, pe,
                                                             WQ, WP, PE);
  // 2) QKV projection: N=1152 (9 tiles/wave), fp32 input, bf16 qkv output
  gemm_kernel<9, true, true><<<4096, 256, 0, stream>>>(
      (const void*)x, WQ, bqkv, QB, KB, VB, nullptr);
  // 3) windowed attention (WMMA qpe + streaming softmax)
  attn_kernel<<<(B_ * HEADS_ * S_) / 128, 128, 0, stream>>>(QB, KB, VB, PE, AO);
  // 4) output projection: N=384 (3 tiles/wave), bf16 input, fp32 output
  gemm_kernel<3, false, false><<<4096, 256, 0, stream>>>(
      (const void*)AO, WP, bproj, nullptr, nullptr, nullptr, (float*)d_out);
}
